// CasualBlock_26568667693691
// MI455X (gfx1250) — matrix operand order repaired and checked by reading, not compiled, not run on hardware
//
#include <hip/hip_runtime.h>
#include <hip/hip_bf16.h>

// ---------------------------------------------------------------------------
// Types / helpers for CDNA5 WMMA (gfx1250, wave32)
// ---------------------------------------------------------------------------
typedef __attribute__((ext_vector_type(16))) __bf16 bf16x16;
typedef __attribute__((ext_vector_type(8)))  float  f32x8;

union FragAB {
    bf16x16        v;
    unsigned short u16[16];
    unsigned int   u32[8];
};
union AccC {
    f32x8 v;
    float f[8];
};

__device__ __forceinline__ unsigned short f2bf(float x) {
    unsigned int u = __float_as_uint(x);
    unsigned int r = u + 0x7FFFu + ((u >> 16) & 1u);   // round-to-nearest-even
    return (unsigned short)(r >> 16);
}

// A-matrix (16x32 bf16) per-lane fragment from row-major LDS row pointer.
// VGPR p holds K pair base = 2p + 16*half (contiguous, matches B), lane M = lane&15.
__device__ __forceinline__ FragAB load_a_frag(const unsigned short* rowp, int half) {
    FragAB f;
#pragma unroll
    for (int p = 0; p < 8; ++p) {
        int k0 = 2 * p + half * 16;
        f.u32[p] = *(const unsigned int*)(rowp + k0);
    }
    return f;
}

// B-matrix (32x16 bf16) fragment: lane holds 16 contiguous K values of column
// N = lane&15; lanes 16-31 hold K = 16..31.  p16 points at Bt[N][16*half].
__device__ __forceinline__ FragAB load_b_frag(const unsigned short* p16) {
    FragAB f;
    const uint4* q = (const uint4*)p16;
    uint4 a = q[0], b = q[1];
    f.u32[0] = a.x; f.u32[1] = a.y; f.u32[2] = a.z; f.u32[3] = a.w;
    f.u32[4] = b.x; f.u32[5] = b.y; f.u32[6] = b.z; f.u32[7] = b.w;
    return f;
}

__device__ __forceinline__ f32x8 wmma_bf16(const FragAB& a, const FragAB& b, f32x8 c) {
    return __builtin_amdgcn_wmma_f32_16x16x32_bf16(
        false, a.v, false, b.v, (short)0, c, false, false);
}

// Async global->LDS copy, 16B per lane.  Per ISA, offset:IMM is added to BOTH
// the LDS destination (VDST) and the global address, so one base register pair
// serves several copies (no per-copy VALU address math).
template <int OFF>
__device__ __forceinline__ void async_copy_b128o(unsigned lds_off, const void* gptr) {
    asm volatile("global_load_async_to_lds_b128 %0, %1, off offset:%2"
                 :: "v"(lds_off), "v"(gptr), "i"(OFF) : "memory");
}
__device__ __forceinline__ void wait_async0() {
    asm volatile("s_wait_asynccnt 0x0" ::: "memory");
}
__device__ __forceinline__ unsigned lds_addr(const void* p) {
    return (unsigned)(uintptr_t)p;   // low 32 bits of generic shared ptr = LDS addr
}

__device__ __forceinline__ float gelu_tanh(float v) {
    const float k = 0.7978845608028654f;   // sqrt(2/pi)
    return 0.5f * v * (1.0f + tanhf(k * (v + 0.044715f * v * v * v)));
}

// ---------------------------------------------------------------------------
// fp32 -> bf16 conversion (weights)
// ---------------------------------------------------------------------------
__global__ void k_f2bf(const float* __restrict__ src, unsigned short* __restrict__ dst, long n) {
    long i = (long)blockIdx.x * 256 + threadIdx.x;
    long stride = (long)gridDim.x * 256;
    for (; i < n; i += stride) dst[i] = f2bf(src[i]);
}

// ---------------------------------------------------------------------------
// LayerNorm over D=1024, one block (256 thr) per row, bf16 output
// ---------------------------------------------------------------------------
__global__ void k_ln(const float* __restrict__ x, const float* __restrict__ w,
                     unsigned short* __restrict__ out) {
    const int D = 1024;
    int row = blockIdx.x;
    int t = threadIdx.x;
    const float* xr = x + (size_t)row * D;
    __shared__ float red[256];

    float v[4];
    float s = 0.0f;
#pragma unroll
    for (int i = 0; i < 4; ++i) { v[i] = xr[t + i * 256]; s += v[i]; }
    red[t] = s; __syncthreads();
    for (int o = 128; o > 0; o >>= 1) { if (t < o) red[t] += red[t + o]; __syncthreads(); }
    float mu = red[0] * (1.0f / D);
    __syncthreads();

    float s2 = 0.0f;
#pragma unroll
    for (int i = 0; i < 4; ++i) { float d = v[i] - mu; s2 += d * d; }
    red[t] = s2; __syncthreads();
    for (int o = 128; o > 0; o >>= 1) { if (t < o) red[t] += red[t + o]; __syncthreads(); }
    float inv = rsqrtf(red[0] * (1.0f / D) + 1e-5f);

#pragma unroll
    for (int i = 0; i < 4; ++i) {
        int c = t + i * 256;
        out[(size_t)row * D + c] = f2bf((v[i] - mu) * inv * w[c]);
    }
}

// ---------------------------------------------------------------------------
// Tiled WMMA GEMM: C[M,N] = A[M,K](bf16) * W[K,N](bf16) + bias
//   EPI=0: +bias -> bf16 out
//   EPI=1: +bias +res(f32) -> f32 out
//   EPI=2: +bias, GELU -> bf16 out
// Block: 256 threads (8 waves). Block tile 128x128, wave tile 32x64.
// K staged in 64-wide chunks, double-buffered LDS (64 KB), A tile via async
// global->LDS (offset-folded), W tile transposed scatter.
// 16 WMMAs per barrier interval per wave.
// ---------------------------------------------------------------------------
__device__ __forceinline__ void gemm_preload(const unsigned short* __restrict__ A,
                                             const unsigned short* __restrict__ W,
                                             unsigned short* bufA, unsigned short* bufBt,
                                             int m0, int n0, int kt, int Kdim, int Ndim,
                                             int t) {
    // A tile: 128 rows x 64 k; two threads per row, 32 bf16 (64B) each -> 4x b128
    int row = t >> 1, ks = (t & 1) * 32;
    const unsigned short* g = A + (size_t)(m0 + row) * Kdim + kt + ks;
    unsigned lo = lds_addr(bufA + row * 64 + ks);
    async_copy_b128o<0>(lo, g);
    async_copy_b128o<16>(lo, g);
    async_copy_b128o<32>(lo, g);
    async_copy_b128o<48>(lo, g);
    if (kt + 128 < Kdim)
        __builtin_prefetch(g + 128, 0, 0);         // global_prefetch_b8
    // W tile transposed: [k][n] -> [n][k] scatter (64 x 128 elements)
#pragma unroll 8
    for (int i = 0; i < 32; ++i) {
        int idx = t * 32 + i;
        int kk = idx >> 7, nn = idx & 127;
        bufBt[nn * 64 + kk] = W[(size_t)(kt + kk) * Ndim + n0 + nn];
    }
}

template <int EPI>
__global__ __launch_bounds__(256) void k_gemm(const unsigned short* __restrict__ A,
                                              const unsigned short* __restrict__ W,
                                              const float* __restrict__ bias,
                                              const float* res,
                                              float* outF,
                                              unsigned short* outB,
                                              int Kdim, int Ndim) {
    __shared__ unsigned short ldsA[2][128 * 64];   // [m][k]
    __shared__ unsigned short ldsBt[2][128 * 64];  // [n][k]

    int t = threadIdx.x;
    int lane = t & 31, w = t >> 5;
    int half = lane >> 4, ln = lane & 15;
    int n0 = blockIdx.x * 128, m0 = blockIdx.y * 128;
    int wm0 = (w >> 1) * 32, wn0 = (w & 1) * 64;

    AccC acc[2][4] = {};

    gemm_preload(A, W, ldsA[0], ldsBt[0], m0, n0, 0, Kdim, Ndim, t);
    wait_async0();
    __syncthreads();

    int buf = 0;
    for (int kt = 0; kt < Kdim; kt += 64) {
        if (kt + 64 < Kdim)
            gemm_preload(A, W, ldsA[buf ^ 1], ldsBt[buf ^ 1], m0, n0, kt + 64, Kdim, Ndim, t);

#pragma unroll
        for (int ks = 0; ks < 64; ks += 32) {
            FragAB aF[2], bF[4];
#pragma unroll
            for (int mt = 0; mt < 2; ++mt)
                aF[mt] = load_a_frag(ldsA[buf] + (wm0 + mt * 16 + ln) * 64 + ks, half);
#pragma unroll
            for (int nt = 0; nt < 4; ++nt)
                bF[nt] = load_b_frag(ldsBt[buf] + (wn0 + nt * 16 + ln) * 64 + ks + half * 16);
#pragma unroll
            for (int mt = 0; mt < 2; ++mt)
#pragma unroll
                for (int nt = 0; nt < 4; ++nt)
                    acc[mt][nt].v = wmma_bf16(aF[mt], bF[nt], acc[mt][nt].v);
        }

        wait_async0();       // next chunk's async A copies landed in LDS
        __syncthreads();     // + dscnt for Bt stores / frag reads, publish buffers
        buf ^= 1;
    }

    // Epilogue: C layout -> VGPR i holds row (i + 8*half), col = ln within tile
#pragma unroll
    for (int mt = 0; mt < 2; ++mt)
#pragma unroll
        for (int nt = 0; nt < 4; ++nt)
#pragma unroll
            for (int i = 0; i < 8; ++i) {
                int r = m0 + wm0 + mt * 16 + i + 8 * half;
                int c = n0 + wn0 + nt * 16 + ln;
                size_t idx = (size_t)r * Ndim + c;
                float v = acc[mt][nt].f[i] + bias[c];
                if (EPI == 0) {
                    outB[idx] = f2bf(v);
                } else if (EPI == 1) {
                    outF[idx] = v + res[idx];
                } else {
                    outB[idx] = f2bf(gelu_tanh(v));
                }
            }
}

// ---------------------------------------------------------------------------
// Causal flash attention.  Grid: (T/64, B*H).  Block: 128 threads (4 waves).
// qkv: bf16 [B,T,3C] (q | k | v), head dim 64.  scale = 1/sqrt(B) (faithful).
// ---------------------------------------------------------------------------
__global__ __launch_bounds__(128) void k_attn(const unsigned short* __restrict__ qkv,
                                              unsigned short* __restrict__ obf) {
    const int T = 2048, C = 1024, C3 = 3072;
    const float scale = 0.70710678118654752f;  // 1/sqrt(B=2)

    int i0 = blockIdx.x * 64;
    int bh = blockIdx.y;
    int b = bh >> 4, h = bh & 15;

    __shared__ unsigned short ldsQ[64 * 64];    // [qrow][d]
    __shared__ unsigned short ldsK[64 * 64];    // [key][d]  (== Bt for S=Q*K^T)
    __shared__ unsigned short ldsVt[64 * 64];   // [d][key]  (== Bt for O=P*V)
    __shared__ unsigned short ldsP[4][16 * 64]; // per-wave P (bf16) [row][key]
    __shared__ float ldsS[4][16 * 64];          // per-wave S (f32)  [row][key]
    __shared__ float ldsRow[4][16];             // per-wave alpha / l

    int t = threadIdx.x;
    int lane = t & 31, w = t >> 5;
    int half = lane >> 4, ln = lane & 15;

    // Q tile (64 x 64) via async global->LDS: 4x b128 per thread, offset-folded
    {
        int row = t >> 1, seg = (t & 1) * 32;
        const unsigned short* g = qkv + (size_t)(b * T + i0 + row) * C3 + h * 64 + seg;
        unsigned lo = lds_addr(ldsQ + row * 64 + seg);
        async_copy_b128o<0>(lo, g);
        async_copy_b128o<16>(lo, g);
        async_copy_b128o<32>(lo, g);
        async_copy_b128o<48>(lo, g);
    }

    AccC accO[4] = {};
    float mprev = -__builtin_inff();
    float lsum = 0.0f;
    int qr = i0 + w * 16 + ln;     // query row this lane handles in softmax

    int nblk = blockIdx.x + 1;     // causal: only key blocks j0 <= i0
    for (int jb = 0; jb < nblk; ++jb) {
        __syncthreads();
        // K tile via async global->LDS
        {
            int row = t >> 1, seg = (t & 1) * 32;
            const unsigned short* g = qkv + (size_t)(b * T + jb * 64 + row) * C3 + C + h * 64 + seg;
            unsigned lo = lds_addr(ldsK + row * 64 + seg);
            async_copy_b128o<0>(lo, g);
            async_copy_b128o<16>(lo, g);
            async_copy_b128o<32>(lo, g);
            async_copy_b128o<48>(lo, g);
        }
        // V tile, transposed into [d][key]
#pragma unroll 4
        for (int i = 0; i < 32; ++i) {
            int idx = t * 32 + i;
            int key = idx >> 6, dd = idx & 63;
            ldsVt[dd * 64 + key] = qkv[(size_t)(b * T + jb * 64 + key) * C3 + 2 * C + h * 64 + dd];
        }
        wait_async0();     // Q (first pass) + K landed in LDS
        __syncthreads();

        // S = Q * K^T for this wave's 16 rows (4 n-tiles x 2 k-steps of 32)
#pragma unroll
        for (int nt = 0; nt < 4; ++nt) {
            AccC s = {};
#pragma unroll
            for (int ks = 0; ks < 64; ks += 32) {
                FragAB aF = load_a_frag(ldsQ + (w * 16 + ln) * 64 + ks, half);
                FragAB bF = load_b_frag(ldsK + (nt * 16 + ln) * 64 + ks + half * 16);
                s.v = wmma_bf16(aF, bF, s.v);
            }
#pragma unroll
            for (int i = 0; i < 8; ++i)
                ldsS[w][(i + 8 * half) * 64 + nt * 16 + ln] = s.f[i];
        }
        __syncthreads();

        // Online softmax: lane ln owns row ln, lane half picks 32-key segment
        int kbase = jb * 64 + half * 32;
        float mx = -__builtin_inff();
        for (int j = 0; j < 32; ++j) {
            float s = ldsS[w][ln * 64 + half * 32 + j] * scale;
            if (kbase + j > qr) s = -__builtin_inff();
            mx = fmaxf(mx, s);
        }
        mx = fmaxf(mx, __shfl_xor(mx, 16, 32));
        float mnew = fmaxf(mprev, mx);
        float sum = 0.0f;
        for (int j = 0; j < 32; ++j) {
            float s = ldsS[w][ln * 64 + half * 32 + j] * scale;
            float p = (kbase + j > qr) ? 0.0f : expf(s - mnew);
            sum += p;
            ldsP[w][ln * 64 + half * 32 + j] = f2bf(p);
        }
        sum += __shfl_xor(sum, 16, 32);
        float alpha = expf(mprev - mnew);
        lsum = lsum * alpha + sum;
        mprev = mnew;
        if (half == 0) ldsRow[w][ln] = alpha;
        __syncthreads();

        // Rescale accumulated O, then O += P * V
#pragma unroll
        for (int nt = 0; nt < 4; ++nt)
#pragma unroll
            for (int i = 0; i < 8; ++i)
                accO[nt].f[i] *= ldsRow[w][i + 8 * half];
#pragma unroll
        for (int nt = 0; nt < 4; ++nt) {
#pragma unroll
            for (int ks = 0; ks < 64; ks += 32) {
                FragAB aF = load_a_frag(ldsP[w] + ln * 64 + ks, half);
                FragAB bF = load_b_frag(ldsVt + (nt * 16 + ln) * 64 + ks + half * 16);
                accO[nt].v = wmma_bf16(aF, bF, accO[nt].v);
            }
        }
    }

    // Finalize: divide by l, store bf16 to o[B,T,C]
    __syncthreads();
    if (half == 0) ldsRow[w][ln] = lsum;
    __syncthreads();
#pragma unroll
    for (int nt = 0; nt < 4; ++nt) {
#pragma unroll
        for (int i = 0; i < 8; ++i) {
            int r = i0 + w * 16 + i + 8 * half;
            float v = accO[nt].f[i] / ldsRow[w][i + 8 * half];
            obf[(size_t)(b * T + r) * C + h * 64 + nt * 16 + ln] = f2bf(v);
        }
    }
}

// ---------------------------------------------------------------------------
// Orchestration
// ---------------------------------------------------------------------------
extern "C" void kernel_launch(void* const* d_in, const int* in_sizes, int n_in,
                              void* d_out, int out_size, void* d_ws, size_t ws_size,
                              hipStream_t stream) {
    (void)in_sizes; (void)n_in; (void)out_size; (void)ws_size;

    const float* x      = (const float*)d_in[0];
    const float* ln1_w  = (const float*)d_in[1];
    const float* attn_w = (const float*)d_in[2];
    const float* attn_b = (const float*)d_in[3];
    const float* proj_w = (const float*)d_in[4];
    const float* proj_b = (const float*)d_in[5];
    const float* ln2_w  = (const float*)d_in[6];
    const float* li1_w  = (const float*)d_in[7];
    const float* li1_b  = (const float*)d_in[8];
    const float* li2_w  = (const float*)d_in[9];
    const float* li2_b  = (const float*)d_in[10];
    float* out = (float*)d_out;

    const long D = 1024, C3 = 3072, F = 4096;
    const long Mrows = 4096;  // B*T = 2*2048

    // Workspace layout (bf16 buffers as ushort)
    unsigned short* wA  = (unsigned short*)d_ws;       // attn_w bf16 [1024,3072]
    unsigned short* wP  = wA  + D * C3;                // proj_w bf16 [1024,1024]
    unsigned short* w1  = wP  + D * D;                 // li1_w  bf16 [1024,4096]
    unsigned short* w2  = w1  + D * F;                 // li2_w  bf16 [4096,1024]
    unsigned short* h1  = w2  + F * D;                 // ln1 out bf16 [4096,1024]
    unsigned short* qkv = h1  + Mrows * D;             // qkv bf16 [4096,3072]
    unsigned short* ob  = qkv + Mrows * C3;            // attn out bf16 [4096,1024]
    unsigned short* h2  = ob  + Mrows * D;             // ln2 out bf16 [4096,1024]
    unsigned short* u   = h2  + Mrows * D;             // mlp mid bf16 [4096,4096]

    // Weight conversions fp32 -> bf16
    {
        long n;
        n = D * C3; k_f2bf<<<(int)((n + 255) / 256), 256, 0, stream>>>(attn_w, wA, n);
        n = D * D;  k_f2bf<<<(int)((n + 255) / 256), 256, 0, stream>>>(proj_w, wP, n);
        n = D * F;  k_f2bf<<<(int)((n + 255) / 256), 256, 0, stream>>>(li1_w,  w1, n);
        n = F * D;  k_f2bf<<<(int)((n + 255) / 256), 256, 0, stream>>>(li2_w,  w2, n);
    }

    // LN1
    k_ln<<<(int)Mrows, 256, 0, stream>>>(x, ln1_w, h1);

    // qkv = h1 @ attn_w + attn_b          [4096,1024]x[1024,3072] -> bf16
    k_gemm<0><<<dim3(3072 / 128, 4096 / 128), 256, 0, stream>>>(
        h1, wA, attn_b, nullptr, nullptr, qkv, 1024, 3072);

    // Causal attention -> ob (bf16)
    k_attn<<<dim3(32, 32), 128, 0, stream>>>(qkv, ob);

    // out = x + ob @ proj_w + proj_b       -> f32 residual stream in d_out
    k_gemm<1><<<dim3(1024 / 128, 4096 / 128), 256, 0, stream>>>(
        ob, wP, proj_b, x, out, nullptr, 1024, 1024);

    // LN2 on residual stream
    k_ln<<<(int)Mrows, 256, 0, stream>>>(out, ln2_w, h2);

    // u = gelu(h2 @ li1_w + li1_b)         -> bf16
    k_gemm<2><<<dim3(4096 / 128, 4096 / 128), 256, 0, stream>>>(
        h2, w1, li1_b, nullptr, nullptr, u, 1024, 4096);

    // out = out + u @ li2_w + li2_b        (in-place residual)
    k_gemm<1><<<dim3(1024 / 128, 4096 / 128), 256, 0, stream>>>(
        u, w2, li2_b, out, out, nullptr, 4096, 1024);
}
